// MultiHeadAttention_53206054862850
// MI455X (gfx1250) — compile-verified, matrix operand order checked by reading, not run on hardware
//
#include <hip/hip_runtime.h>
#include <hip/hip_bf16.h>

typedef __attribute__((ext_vector_type(16))) _Float16 v16h;
typedef __attribute__((ext_vector_type(8)))  _Float16 v8h;
typedef __attribute__((ext_vector_type(8)))  float    v8f;

#define TSEQ   4096
#define NHEAD  16
#define DK     64
#define DMODEL 1024

#define WMMA_F32_F16(a, b, c) \
  __builtin_amdgcn_wmma_f32_16x16x32_f16(false, (a), false, (b), (short)0, (c), false, false)

// ---------------------------------------------------------------------------
// Async global->LDS copy (CDNA5 GLOBAL_LOAD_ASYNC_TO_LDS_B128, ASYNCcnt-tracked)
// via inline asm — portable across toolchains. vdst = wave-relative LDS byte
// offset = low 32 bits of the generic pointer.
// ---------------------------------------------------------------------------
static __device__ __forceinline__ void cp16(void* l, const void* g) {
  unsigned           lds = (unsigned)(size_t)l;
  unsigned long long ga  = (unsigned long long)(size_t)g;
  asm volatile("global_load_async_to_lds_b128 %0, %1, off"
               :: "v"(lds), "v"(ga) : "memory");
}
#define WAIT_ASYNC(n) asm volatile("s_wait_asynccnt " #n ::: "memory")

// Load a 16-bit A/B WMMA fragment: per lane, 8 contiguous halves at p and 8 at p+16
// (CDNA5 16-bit layout: elems 0..7 -> K = h*8+0..7, elems 8..15 -> K = 16+h*8+0..7).
static __device__ __forceinline__ v16h ld_frag(const _Float16* p) {
  v8h lo = *(const v8h*)p;
  v8h hi = *(const v8h*)(p + 16);
  v16h r;
#pragma unroll
  for (int i = 0; i < 8; ++i) { r[i] = lo[i]; r[i + 8] = hi[i]; }
  return r;
}

// Row-wise reductions across the 16 lanes that share a C-matrix row (wave32).
static __device__ __forceinline__ float rowmax16(float x) {
#pragma unroll
  for (int off = 1; off < 16; off <<= 1) x = fmaxf(x, __shfl_xor(x, off, 32));
  return x;
}
static __device__ __forceinline__ float rowsum16(float x) {
#pragma unroll
  for (int off = 1; off < 16; off <<= 1) x += __shfl_xor(x, off, 32);
  return x;
}

// ---------------------------------------------------------------------------
// f16 GEMM: C[4096][1024] = A[4096][1024] * BT[1024][1024]^T + bias.
// Block = 128 thr = 4 waves; tile 128x64, K staged 64 at a time in LDS (48 KB,
// double-buffered async). Each wave: 32x64 strip = 16 WMMA per stage, with all
// fragments loaded into distinct registers before the WMMA burst.
// ---------------------------------------------------------------------------
__global__ __launch_bounds__(128) void gemm_f16_kernel(
    const _Float16* __restrict__ A,
    const _Float16* __restrict__ BT,
    const float*    __restrict__ bias,
    float*          __restrict__ Cf,
    _Float16*       __restrict__ Ch)
{
  constexpr int N = DMODEL, K = DMODEL, BK = 64;
  __shared__ _Float16 aT[2][128 * BK];   // 16 KB each: [row][k], row stride 64 halves
  __shared__ _Float16 bT[2][64 * BK];    //  8 KB each: [col][k]

  const int tid  = threadIdx.x;          // 0..127
  const int lane = tid & 31;
  const int wave = tid >> 5;             // 0..3
  const int lr = lane & 15;
  const int lh = lane >> 4;
  const int rowBase = blockIdx.y * 128;
  const int colBase = blockIdx.x * 64;

  // Copy one K-stage: A 128x128B (1024 chunks -> 8/thr), B 64x128B (512 -> 4/thr).
  auto copy_stage = [&](int k0, int buf) {
#pragma unroll
    for (int i = 0; i < 8; ++i) {
      const int chunk = tid + i * 128;
      const int r = chunk >> 3, c = chunk & 7;
      cp16(&aT[buf][r * BK + c * 8], A + (size_t)(rowBase + r) * K + k0 + c * 8);
    }
#pragma unroll
    for (int i = 0; i < 4; ++i) {
      const int chunk = tid + i * 128;
      const int r = chunk >> 3, c = chunk & 7;
      cp16(&bT[buf][r * BK + c * 8], BT + (size_t)(colBase + r) * K + k0 + c * 8);
    }
  };

  v8f acc[2][4] = {};

  copy_stage(0, 0);
  constexpr int NST = K / BK;
  for (int st = 0; st < NST; ++st) {
    const int buf = st & 1;
    if (st + 1 < NST) { copy_stage((st + 1) * BK, (st + 1) & 1); WAIT_ASYNC(12); }
    else              { WAIT_ASYNC(0); }
    __syncthreads();

    // Batch all fragment loads (distinct registers), then the 16-WMMA burst.
    v16h a[2][2], b[4][2];
#pragma unroll
    for (int mt = 0; mt < 2; ++mt) {
      const _Float16* arow = &aT[buf][(wave * 32 + mt * 16 + lr) * BK + lh * 8];
      a[mt][0] = ld_frag(arow);
      a[mt][1] = ld_frag(arow + 32);
    }
#pragma unroll
    for (int nt = 0; nt < 4; ++nt) {
      const _Float16* brow = &bT[buf][(nt * 16 + lr) * BK + lh * 8];
      b[nt][0] = ld_frag(brow);
      b[nt][1] = ld_frag(brow + 32);
    }
#pragma unroll
    for (int mt = 0; mt < 2; ++mt)
#pragma unroll
      for (int nt = 0; nt < 4; ++nt) {
        acc[mt][nt] = WMMA_F32_F16(a[mt][0], b[nt][0], acc[mt][nt]);
        acc[mt][nt] = WMMA_F32_F16(a[mt][1], b[nt][1], acc[mt][nt]);
      }
    __syncthreads();
  }

#pragma unroll
  for (int mt = 0; mt < 2; ++mt)
#pragma unroll
    for (int nt = 0; nt < 4; ++nt) {
      const int col = colBase + nt * 16 + lr;
      const float bv = bias[col];
#pragma unroll
      for (int j = 0; j < 8; ++j) {
        const int row = rowBase + wave * 32 + mt * 16 + lh * 8 + j;
        const float v = acc[mt][nt][j] + bv;
        if (Ch) Ch[(size_t)row * N + col] = (_Float16)v;
        else    Cf[(size_t)row * N + col] = v;
      }
    }
}

// ---------------------------------------------------------------------------
// Flash attention per head. Grid: (T/128, H); 8 waves x 16 q-rows.
// K/V tiles (64 keys) staged in LDS, double-buffered async; P tile goes
// C-layout -> LDS -> A-layout for the P*V WMMA.
// ---------------------------------------------------------------------------
__global__ __launch_bounds__(256) void attn_kernel(
    const _Float16* __restrict__ Qh,    // [H][T][64], pre-scaled by 1/8
    const _Float16* __restrict__ Kh,    // [H][T][64]
    const _Float16* __restrict__ VhT,   // [H][64][T]
    _Float16*       __restrict__ attnh) // [T][H*64]
{
  constexpr int KT = 64;                      // keys per stage
  __shared__ _Float16 kT[2][KT * DK];         // 8 KB each: [key][kd]
  __shared__ _Float16 vT[2][DK * KT];         // 8 KB each: [n][key]
  __shared__ _Float16 plds[8][16][32];        // 8 KB: per-wave P staging

  const int tid  = threadIdx.x;
  const int lane = tid & 31;
  const int wave = tid >> 5;
  const int lr = lane & 15;
  const int lh = lane >> 4;
  const int h = blockIdx.y;
  const int qBase = blockIdx.x * 128 + wave * 16;

  const _Float16* Khead = Kh  + (size_t)h * TSEQ * DK;
  const _Float16* Vhead = VhT + (size_t)h * DK * TSEQ;

  const _Float16* Qp = Qh + ((size_t)h * TSEQ + qBase + lr) * DK + lh * 8;
  v16h qa0 = ld_frag(Qp);
  v16h qa1 = ld_frag(Qp + 32);

  // Copy one 64-key stage: K 64x128B (512 chunks -> 2/thr), V 64x128B (512 -> 2/thr).
  auto copy_stage = [&](int kb, int buf) {
#pragma unroll
    for (int i = 0; i < 2; ++i) {
      const int chunk = tid + i * 256;
      const int r = chunk >> 3, c = chunk & 7;
      cp16(&kT[buf][r * DK + c * 8], Khead + (size_t)(kb + r) * DK + c * 8);
    }
#pragma unroll
    for (int i = 0; i < 2; ++i) {
      const int chunk = tid + i * 256;
      const int r = chunk >> 3, c = chunk & 7;
      cp16(&vT[buf][r * KT + c * 8], Vhead + (size_t)r * TSEQ + kb + c * 8);
    }
  };

  v8f o[4] = {};
  float m[8], l[8];
#pragma unroll
  for (int j = 0; j < 8; ++j) { m[j] = -1e30f; l[j] = 0.0f; }

  copy_stage(0, 0);
  constexpr int NST = TSEQ / KT;
  for (int st = 0; st < NST; ++st) {
    const int buf = st & 1;
    if (st + 1 < NST) { copy_stage((st + 1) * KT, (st + 1) & 1); WAIT_ASYNC(4); }
    else              { WAIT_ASYNC(0); }
    __syncthreads();

#pragma unroll
    for (int sub = 0; sub < 2; ++sub) {           // two 32-key steps per stage
      const int kb = sub * 32;
      const _Float16* kp0 = &kT[buf][(kb + lr) * DK + lh * 8];
      const _Float16* kp1 = &kT[buf][(kb + 16 + lr) * DK + lh * 8];
      v16h kf[2][2];
      kf[0][0] = ld_frag(kp0);
      kf[0][1] = ld_frag(kp0 + 32);
      kf[1][0] = ld_frag(kp1);
      kf[1][1] = ld_frag(kp1 + 32);

      v8f s0 = {}, s1 = {};
      s0 = WMMA_F32_F16(qa0, kf[0][0], s0);
      s0 = WMMA_F32_F16(qa1, kf[0][1], s0);
      s1 = WMMA_F32_F16(qa0, kf[1][0], s1);
      s1 = WMMA_F32_F16(qa1, kf[1][1], s1);

      // Online softmax; slot j <-> row j + 8*lh
#pragma unroll
      for (int j = 0; j < 8; ++j) {
        float mx = rowmax16(fmaxf(s0[j], s1[j]));
        float mn = fmaxf(m[j], mx);
        float alpha = __expf(m[j] - mn);
        m[j] = mn;
        float p0 = __expf(s0[j] - mn);
        float p1 = __expf(s1[j] - mn);
        s0[j] = p0; s1[j] = p1;
        l[j] = l[j] * alpha + rowsum16(p0 + p1);
#pragma unroll
        for (int nt = 0; nt < 4; ++nt) o[nt][j] *= alpha;
      }

      __syncthreads();
#pragma unroll
      for (int j = 0; j < 8; ++j) {
        const int r = lh * 8 + j;
        plds[wave][r][lr]      = (_Float16)s0[j];
        plds[wave][r][16 + lr] = (_Float16)s1[j];
      }
      __syncthreads();

      // Batch pa + all V fragments, then the 4-WMMA burst.
      v16h pa = ld_frag(&plds[wave][lr][lh * 8]);
      v16h vf[4];
#pragma unroll
      for (int nt = 0; nt < 4; ++nt)
        vf[nt] = ld_frag(&vT[buf][(nt * 16 + lr) * KT + kb + lh * 8]);
#pragma unroll
      for (int nt = 0; nt < 4; ++nt)
        o[nt] = WMMA_F32_F16(pa, vf[nt], o[nt]);
    }
    __syncthreads();
  }

#pragma unroll
  for (int j = 0; j < 8; ++j) {
    const float inv = 1.0f / l[j];
    const int row = qBase + lh * 8 + j;
    _Float16* outp = attnh + (size_t)row * DMODEL + h * DK;
#pragma unroll
    for (int nt = 0; nt < 4; ++nt)
      outp[nt * 16 + lr] = (_Float16)(o[nt][j] * inv);
  }
}

// ---------------------------------------------------------------------------
// Small conversion / layout kernels
// ---------------------------------------------------------------------------
__global__ void cvt_f32_f16(const float* __restrict__ in, _Float16* __restrict__ out, int n) {
  int i = blockIdx.x * blockDim.x + threadIdx.x;
  if (i < n) out[i] = (_Float16)in[i];
}

// W[K][N] fp32 -> WT[N][K] f16
__global__ void cvt_transpose_w(const float* __restrict__ W, _Float16* __restrict__ WT) {
  int idx = blockIdx.x * blockDim.x + threadIdx.x;     // N*K
  int n = idx >> 10, k = idx & (DMODEL - 1);
  WT[idx] = (_Float16)W[(size_t)k * DMODEL + n];
}

// Rotary on Q and K; re-layout [T][H*64] -> [H][T][64]; fold 1/sqrt(dk)=0.125 into Q.
__global__ void rotary_kernel(const _Float16* __restrict__ Qin,
                              const _Float16* __restrict__ Kin,
                              _Float16* __restrict__ Qout,
                              _Float16* __restrict__ Kout) {
  int idx = blockIdx.x * blockDim.x + threadIdx.x;     // T*H*32
  const int i = idx & 31;
  const int h = (idx >> 5) & (NHEAD - 1);
  const int t = idx >> 9;
  const float inv_freq = __powf(10000.0f, -(float)(2 * i) / (float)DK);
  const float f = (float)t * inv_freq;
  float s, c;
  __sincosf(f, &s, &c);
  const size_t ib = (size_t)t * DMODEL + h * DK;
  const size_t ob = ((size_t)h * TSEQ + t) * DK;
  {
    float x1 = (float)Qin[ib + i], x2 = (float)Qin[ib + 32 + i];
    Qout[ob + i]      = (_Float16)((x1 * c - x2 * s) * 0.125f);
    Qout[ob + 32 + i] = (_Float16)((x2 * c + x1 * s) * 0.125f);
  }
  {
    float x1 = (float)Kin[ib + i], x2 = (float)Kin[ib + 32 + i];
    Kout[ob + i]      = (_Float16)(x1 * c - x2 * s);
    Kout[ob + 32 + i] = (_Float16)(x2 * c + x1 * s);
  }
}

// V [T][H*64] -> VhT [H][64][T]
__global__ void cvt_v_transpose(const _Float16* __restrict__ Vh0, _Float16* __restrict__ VhT) {
  int idx = blockIdx.x * blockDim.x + threadIdx.x;     // T*1024
  int t = idx >> 10;
  int cc = idx & (DMODEL - 1);
  int h = cc >> 6, d = cc & 63;
  VhT[((size_t)h * DK + d) * TSEQ + t] = Vh0[idx];
}

// ---------------------------------------------------------------------------
extern "C" void kernel_launch(void* const* d_in, const int* in_sizes, int n_in,
                              void* d_out, int out_size, void* d_ws, size_t ws_size,
                              hipStream_t stream) {
  (void)in_sizes; (void)n_in; (void)out_size; (void)ws_size;
  const float* x  = (const float*)d_in[0];
  const float* Wq = (const float*)d_in[1];
  const float* bq = (const float*)d_in[2];
  const float* Wk = (const float*)d_in[3];
  const float* bk = (const float*)d_in[4];
  const float* Wv = (const float*)d_in[5];
  const float* bv = (const float*)d_in[6];
  const float* Wo = (const float*)d_in[7];
  const float* bo = (const float*)d_in[8];
  float* out = (float*)d_out;

  char* w = (char*)d_ws;
  const size_t MB = (size_t)1 << 20;
  _Float16* xh    = (_Float16*)(w + 0);        //  8 MB  x as f16 [T][D]
  _Float16* WqT   = (_Float16*)(w + 8  * MB);  //  2 MB  each, transposed f16
  _Float16* WkT   = (_Float16*)(w + 10 * MB);
  _Float16* WvT   = (_Float16*)(w + 12 * MB);
  _Float16* WoT   = (_Float16*)(w + 14 * MB);
  _Float16* Qh0   = (_Float16*)(w + 16 * MB);  //  8 MB  pre-rotary Q [T][D]
  _Float16* Kh0   = (_Float16*)(w + 24 * MB);
  _Float16* Vh0   = (_Float16*)(w + 32 * MB);
  _Float16* Qh    = (_Float16*)(w + 40 * MB);  //  8 MB  [H][T][64], scaled
  _Float16* Kh    = (_Float16*)(w + 48 * MB);  //  8 MB  [H][T][64]
  _Float16* VhT   = (_Float16*)(w + 56 * MB);  //  8 MB  [H][64][T]
  _Float16* attnh = (_Float16*)(w + 64 * MB);  //  8 MB  [T][D]

  cvt_f32_f16<<<(TSEQ * DMODEL) / 256, 256, 0, stream>>>(x, xh, TSEQ * DMODEL);
  cvt_transpose_w<<<(DMODEL * DMODEL) / 256, 256, 0, stream>>>(Wq, WqT);
  cvt_transpose_w<<<(DMODEL * DMODEL) / 256, 256, 0, stream>>>(Wk, WkT);
  cvt_transpose_w<<<(DMODEL * DMODEL) / 256, 256, 0, stream>>>(Wv, WvT);
  cvt_transpose_w<<<(DMODEL * DMODEL) / 256, 256, 0, stream>>>(Wo, WoT);

  dim3 gg(DMODEL / 64, TSEQ / 128);
  gemm_f16_kernel<<<gg, 128, 0, stream>>>(xh, WqT, bq, nullptr, Qh0);
  gemm_f16_kernel<<<gg, 128, 0, stream>>>(xh, WkT, bk, nullptr, Kh0);
  gemm_f16_kernel<<<gg, 128, 0, stream>>>(xh, WvT, bv, nullptr, Vh0);

  rotary_kernel<<<(TSEQ * NHEAD * 32) / 256, 256, 0, stream>>>(Qh0, Kh0, Qh, Kh);
  cvt_v_transpose<<<(TSEQ * DMODEL) / 256, 256, 0, stream>>>(Vh0, VhT);

  dim3 ga(TSEQ / 128, NHEAD);
  attn_kernel<<<ga, 256, 0, stream>>>(Qh, Kh, VhT, attnh);

  gemm_f16_kernel<<<gg, 128, 0, stream>>>(attnh, WoT, bo, out, nullptr);
}
